// LaneFusionEncoder_41154376630815
// MI455X (gfx1250) — matrix operand order repaired and checked by reading, not compiled, not run on hardware
//
#include <hip/hip_runtime.h>
#include <hip/hip_bf16.h>

typedef _Float16 v16h __attribute__((ext_vector_type(16)));
typedef float    v8f  __attribute__((ext_vector_type(8)));

#define NSEQ (16 * 512)
#define VLEN 20
#define CCH  128
#define TTOK 64
#define HEMB 192
#define NTHREADS 256

struct Params {
    const float *x, *tl_table, *lane_table;
    const float *ch_fc1_w, *ch_fc1_b, *ch_fc2_w, *ch_fc2_b;
    const float *tok_fc1_w, *tok_fc1_b, *tok_fc2_w, *tok_fc2_b;
    const float *blk_norm1_w, *blk_norm1_b;
    const float *blk_tok_fc1_w, *blk_tok_fc1_b, *blk_tok_fc2_w, *blk_tok_fc2_b;
    const float *blk_norm2_w, *blk_norm2_b;
    const float *blk_ch_fc1_w, *blk_ch_fc1_b, *blk_ch_fc2_w, *blk_ch_fc2_b;
    const float *norm_w, *norm_b;
    const float *emb_fc1_w, *emb_fc1_b, *emb_fc2_w, *emb_fc2_b;
    float *out;
};

__device__ __forceinline__ float gelu_exact(float x) {
    return 0.5f * x * (1.0f + erff(x * 0.70710678118654752440f));
}

// Generic LDS-resident GEMM: C(MxN) = A(MxK) * B(KxN), f16 inputs, f32 accum.
// A layout per CDNA5 16-bit A 16x32 fragment; B per 32x16 fragment; C per f32 16x16.
template <int M, int N, int K>
__device__ __forceinline__ void gemm_lds(const _Float16 *__restrict__ A,
                                         const _Float16 *__restrict__ B,
                                         float *__restrict__ C, int tid) {
    const int lane = tid & 31;
    const int wave = tid >> 5;
    const int ml = lane & 15;   // row/col within 16
    const int hi = lane >> 4;   // lane-group select
    const int nt = N / 16;
    const int ntiles = (M / 16) * nt;
    for (int t = wave; t < ntiles; t += 8) {
        const int tm = (t / nt) * 16;
        const int tn = (t % nt) * 16;
        v8f acc = {};
#pragma unroll
        for (int k0 = 0; k0 < K; k0 += 32) {
            v16h a, b;
            // A fragment: lane group hi selects K-subblock offset 8*hi and 16+8*hi
            const _Float16 *ap = A + (tm + ml) * K + k0 + hi * 8;
#pragma unroll
            for (int i = 0; i < 8; ++i) { a[i] = ap[i]; a[8 + i] = ap[16 + i]; }
            // B fragment: lane group hi selects K rows [8*hi, 8*hi+8) and [16+8*hi, 24+8*hi)
            const _Float16 *bp = B + (k0 + hi * 8) * N + tn + ml;
#pragma unroll
            for (int i = 0; i < 16; ++i) b[i] = bp[(i < 8 ? i : 8 + i) * N];
            acc = __builtin_amdgcn_wmma_f32_16x16x32_f16(
                false, a, false, b, (short)0, acc, false, false);
        }
        // C fragment: VGPR r holds row tm + r + 8*hi, col tn + ml
        float *cp = C + (tm + hi * 8) * N + tn + ml;
#pragma unroll
        for (int r = 0; r < 8; ++r) cp[r * N] = acc[r];
    }
}

// Convert f32 global weights to f16 LDS, zero-padding [count, total).
__device__ __forceinline__ void load_w16(_Float16 *dst, const float *src,
                                         int count, int total, int tid) {
    for (int i = tid; i < total; i += NTHREADS)
        dst[i] = (i < count) ? (_Float16)src[i] : (_Float16)0.0f;
}

__global__ __launch_bounds__(NTHREADS) void lane_fusion_encoder_kernel(Params P_) {
    __shared__ float    fbuf[TTOK * CCH];   // residual stream (f32)      32 KB
    __shared__ float    ybuf[CCH * TTOK];   // GEMM f32 output scratch    32 KB
    __shared__ _Float16 Abuf[CCH * TTOK];   // A-operand staging (f16)    16 KB
    __shared__ _Float16 Bbuf[CCH * CCH];    // weight staging (f16)       32 KB
    __shared__ float    xg[VLEN * 5];
    __shared__ float    geom[VLEN * 4];
    __shared__ int      tli[VLEN], lti[VLEN];
    __shared__ float    smA[CCH];
    __shared__ float    smB[HEMB];
    __shared__ float    stats[2];
    __shared__ int      validFlag;

    const int tid = threadIdx.x;
    const int s   = blockIdx.x;
    const float *xb = P_.x + (size_t)s * (VLEN * 5);

    if (tid == 0) {
        // CDNA5 global prefetch of the large block weights into cache
        __builtin_prefetch(P_.blk_ch_fc1_w, 0, 3);
        __builtin_prefetch(P_.blk_ch_fc2_w, 0, 3);
        validFlag = 0;
    }

    for (int i = tid; i < VLEN * 5; i += NTHREADS) xg[i] = xb[i];
    __syncthreads();

    // geometry features + embedding indices
    if (tid < VLEN) {
        const int v = tid;
        const float hx = xg[v * 5 + 2];
        geom[v * 4 + 0] = xg[v * 5 + 0];
        geom[v * 4 + 1] = xg[v * 5 + 1];
        geom[v * 4 + 2] = cosf(hx);
        geom[v * 4 + 3] = sinf(hx);
        int a = (int)xg[v * 5 + 3];
        tli[v] = min(max(a, 0), 8);
        int b = (int)xg[v * 5 + 4];
        lti[v] = min(max(b, 0), 19);
    }
    __syncthreads();
    if (tid < VLEN) {
        const bool nz = (geom[tid * 4 + 0] != 0.f) || (geom[tid * 4 + 1] != 0.f) ||
                        (geom[tid * 4 + 2] != 0.f) || (geom[tid * 4 + 3] != 0.f);
        if (nz) atomicOr(&validFlag, 1);
    }

    // ---- Stage 1: channel MLP on geom ------------------------------------
    // h1(20,128) = gelu(geom @ ch_fc1 + b1) computed on VALU (K=4), rows 20..31 zero
    for (int i = tid; i < 32 * CCH; i += NTHREADS) {
        const int v = i >> 7, c = i & 127;
        float acc = 0.f;
        if (v < VLEN) {
            acc = P_.ch_fc1_b[c];
#pragma unroll
            for (int j = 0; j < 4; ++j) acc += geom[v * 4 + j] * P_.ch_fc1_w[j * CCH + c];
            acc = gelu_exact(acc);
        }
        Abuf[i] = (_Float16)acc;
    }
    load_w16(Bbuf, P_.ch_fc2_w, CCH * CCH, CCH * CCH, tid);
    __syncthreads();
    gemm_lds<32, CCH, CCH>(Abuf, Bbuf, ybuf, tid);   // (32,128)@(128,128)
    __syncthreads();

    // f(20,128) = ybuf + bias + tables; store transposed f16 (K pad 20->32)
    for (int i = tid; i < CCH * 32; i += NTHREADS) {
        const int c = i >> 5, v = i & 31;
        float val = 0.f;
        if (v < VLEN) {
            val = ybuf[v * CCH + c] + P_.ch_fc2_b[c] +
                  P_.tl_table[tli[v] * CCH + c] + P_.lane_table[lti[v] * CCH + c];
        }
        Abuf[i] = (_Float16)val;  // Abuf[c*32 + v]: A = f^T (128 x 32pad)
    }
    load_w16(Bbuf, P_.tok_fc1_w, VLEN * TTOK, 32 * TTOK, tid);  // rows 20..31 = 0
    __syncthreads();

    // ---- Stage 2: token MLP (20 -> 64 tokens) ----------------------------
    gemm_lds<CCH, TTOK, 32>(Abuf, Bbuf, ybuf, tid);  // (128,32)@(32,64)
    __syncthreads();
    for (int i = tid; i < CCH * TTOK; i += NTHREADS) {
        const int n = i & 63;
        Abuf[i] = (_Float16)gelu_exact(ybuf[i] + P_.tok_fc1_b[n]);
    }
    load_w16(Bbuf, P_.tok_fc2_w, TTOK * TTOK, TTOK * TTOK, tid);
    __syncthreads();
    gemm_lds<CCH, TTOK, TTOK>(Abuf, Bbuf, ybuf, tid);  // (128,64)@(64,64)
    __syncthreads();
    // f(64,128) = transpose(ybuf) + bias (bias indexed by new-token v)
    for (int i = tid; i < TTOK * CCH; i += NTHREADS) {
        const int v = i >> 7, c = i & 127;
        fbuf[i] = ybuf[c * TTOK + v] + P_.tok_fc2_b[v];
    }
    __syncthreads();

    // ---- Stage 3: DEPTH mixer blocks -------------------------------------
    for (int blk = 0; blk < 3; ++blk) {
        // LN1 over channels -> A = y^T (128 x 64), f16
        {
            const float *nw = P_.blk_norm1_w + blk * CCH;
            const float *nb = P_.blk_norm1_b + blk * CCH;
            if (tid < TTOK) {
                const int v = tid;
                float m = 0.f;
                for (int c = 0; c < CCH; ++c) m += fbuf[v * CCH + c];
                m *= (1.f / CCH);
                float var = 0.f;
                for (int c = 0; c < CCH; ++c) {
                    const float d = fbuf[v * CCH + c] - m;
                    var += d * d;
                }
                var *= (1.f / CCH);
                const float rs = rsqrtf(var + 1e-5f);
                for (int c = 0; c < CCH; ++c) {
                    const float yv = (fbuf[v * CCH + c] - m) * rs * nw[c] + nb[c];
                    Abuf[c * TTOK + v] = (_Float16)yv;
                }
            }
        }
        load_w16(Bbuf, P_.blk_tok_fc1_w + blk * TTOK * TTOK, TTOK * TTOK, TTOK * TTOK, tid);
        __syncthreads();
        gemm_lds<CCH, TTOK, TTOK>(Abuf, Bbuf, ybuf, tid);
        __syncthreads();
        for (int i = tid; i < CCH * TTOK; i += NTHREADS) {
            const int n = i & 63;
            Abuf[i] = (_Float16)gelu_exact(ybuf[i] + P_.blk_tok_fc1_b[blk * TTOK + n]);
        }
        load_w16(Bbuf, P_.blk_tok_fc2_w + blk * TTOK * TTOK, TTOK * TTOK, TTOK * TTOK, tid);
        __syncthreads();
        gemm_lds<CCH, TTOK, TTOK>(Abuf, Bbuf, ybuf, tid);
        __syncthreads();
        for (int i = tid; i < TTOK * CCH; i += NTHREADS) {
            const int v = i >> 7, c = i & 127;
            fbuf[i] += ybuf[c * TTOK + v] + P_.blk_tok_fc2_b[blk * TTOK + v];
        }
        __syncthreads();

        // LN2 over channels -> A = y (64 x 128) row-major, f16
        {
            const float *nw = P_.blk_norm2_w + blk * CCH;
            const float *nb = P_.blk_norm2_b + blk * CCH;
            if (tid < TTOK) {
                const int v = tid;
                float m = 0.f;
                for (int c = 0; c < CCH; ++c) m += fbuf[v * CCH + c];
                m *= (1.f / CCH);
                float var = 0.f;
                for (int c = 0; c < CCH; ++c) {
                    const float d = fbuf[v * CCH + c] - m;
                    var += d * d;
                }
                var *= (1.f / CCH);
                const float rs = rsqrtf(var + 1e-5f);
                for (int c = 0; c < CCH; ++c) {
                    const float yv = (fbuf[v * CCH + c] - m) * rs * nw[c] + nb[c];
                    Abuf[v * CCH + c] = (_Float16)yv;
                }
            }
        }
        load_w16(Bbuf, P_.blk_ch_fc1_w + blk * CCH * CCH, CCH * CCH, CCH * CCH, tid);
        __syncthreads();
        gemm_lds<TTOK, CCH, CCH>(Abuf, Bbuf, ybuf, tid);
        __syncthreads();
        for (int i = tid; i < TTOK * CCH; i += NTHREADS) {
            const int c = i & 127;
            Abuf[i] = (_Float16)gelu_exact(ybuf[i] + P_.blk_ch_fc1_b[blk * CCH + c]);
        }
        load_w16(Bbuf, P_.blk_ch_fc2_w + blk * CCH * CCH, CCH * CCH, CCH * CCH, tid);
        __syncthreads();
        gemm_lds<TTOK, CCH, CCH>(Abuf, Bbuf, ybuf, tid);
        __syncthreads();
        for (int i = tid; i < TTOK * CCH; i += NTHREADS) {
            const int c = i & 127;
            fbuf[i] += ybuf[i] + P_.blk_ch_fc2_b[blk * CCH + c];
        }
        __syncthreads();
    }

    // ---- Stage 4: token mean -> LN -> embedding head (VALU) --------------
    if (tid < CCH) {
        float acc = 0.f;
        for (int v = 0; v < TTOK; ++v) acc += fbuf[v * CCH + tid];
        smA[tid] = acc * (1.f / TTOK);
    }
    __syncthreads();
    if (tid == 0) {
        float m = 0.f;
        for (int c = 0; c < CCH; ++c) m += smA[c];
        m *= (1.f / CCH);
        float var = 0.f;
        for (int c = 0; c < CCH; ++c) {
            const float d = smA[c] - m;
            var += d * d;
        }
        var *= (1.f / CCH);
        stats[0] = m;
        stats[1] = rsqrtf(var + 1e-5f);
    }
    __syncthreads();
    if (tid < CCH)
        smA[tid] = (smA[tid] - stats[0]) * stats[1] * P_.norm_w[tid] + P_.norm_b[tid];
    __syncthreads();
    if (tid < HEMB) {
        float acc = P_.emb_fc1_b[tid];
        for (int c = 0; c < CCH; ++c) acc += smA[c] * P_.emb_fc1_w[c * HEMB + tid];
        smB[tid] = gelu_exact(acc);
    }
    __syncthreads();
    const float validf = validFlag ? 1.f : 0.f;
    if (tid < HEMB) {
        float acc = P_.emb_fc2_b[tid];
        for (int h = 0; h < HEMB; ++h) acc += smB[h] * P_.emb_fc2_w[h * HEMB + tid];
        P_.out[(size_t)s * HEMB + tid] = acc * validf;
    }
    // outputs 2 & 3: mask_p (as 0/1 float) and pos (5 floats)
    if (tid == 0) P_.out[(size_t)NSEQ * HEMB + s] = validFlag ? 0.f : 1.f;
    if (tid < 5) {
        const float mv = xg[(VLEN / 2) * 5 + tid];
        const float scale = (tid == 3) ? (1.f / 8.f) : ((tid == 4) ? (1.f / 19.f) : 1.f);
        P_.out[(size_t)NSEQ * HEMB + NSEQ + (size_t)s * 5 + tid] = mv * scale;
    }
}

extern "C" void kernel_launch(void *const *d_in, const int *in_sizes, int n_in,
                              void *d_out, int out_size, void *d_ws, size_t ws_size,
                              hipStream_t stream) {
    (void)in_sizes; (void)n_in; (void)out_size; (void)d_ws; (void)ws_size;
    Params p;
    p.x             = (const float *)d_in[0];
    p.tl_table      = (const float *)d_in[1];
    p.lane_table    = (const float *)d_in[2];
    p.ch_fc1_w      = (const float *)d_in[3];
    p.ch_fc1_b      = (const float *)d_in[4];
    p.ch_fc2_w      = (const float *)d_in[5];
    p.ch_fc2_b      = (const float *)d_in[6];
    p.tok_fc1_w     = (const float *)d_in[7];
    p.tok_fc1_b     = (const float *)d_in[8];
    p.tok_fc2_w     = (const float *)d_in[9];
    p.tok_fc2_b     = (const float *)d_in[10];
    p.blk_norm1_w   = (const float *)d_in[11];
    p.blk_norm1_b   = (const float *)d_in[12];
    p.blk_tok_fc1_w = (const float *)d_in[13];
    p.blk_tok_fc1_b = (const float *)d_in[14];
    p.blk_tok_fc2_w = (const float *)d_in[15];
    p.blk_tok_fc2_b = (const float *)d_in[16];
    p.blk_norm2_w   = (const float *)d_in[17];
    p.blk_norm2_b   = (const float *)d_in[18];
    p.blk_ch_fc1_w  = (const float *)d_in[19];
    p.blk_ch_fc1_b  = (const float *)d_in[20];
    p.blk_ch_fc2_w  = (const float *)d_in[21];
    p.blk_ch_fc2_b  = (const float *)d_in[22];
    p.norm_w        = (const float *)d_in[23];
    p.norm_b        = (const float *)d_in[24];
    p.emb_fc1_w     = (const float *)d_in[25];
    p.emb_fc1_b     = (const float *)d_in[26];
    p.emb_fc2_w     = (const float *)d_in[27];
    p.emb_fc2_b     = (const float *)d_in[28];
    p.out           = (float *)d_out;

    lane_fusion_encoder_kernel<<<NSEQ, NTHREADS, 0, stream>>>(p);
}
